// UNetDAGs_50929722196482
// MI455X (gfx1250) — matrix operand order repaired and checked by reading, not compiled, not run on hardware
//
#include <hip/hip_runtime.h>

// MI455X / gfx1250 implementation of 6 directional gated scans.
// Strategy: bf16 WMMA (16x16x32, f32 accum) for the per-step C x C matmuls,
// weights held as A-fragments in registers across the whole scan,
// activations round-trip through LDS in B-fragment-friendly [n][k] layout,
// next-step global tiles software-pipelined through registers so HBM latency
// overlaps with the WMMA work of the current step.

typedef __bf16 bf16_t;
typedef __attribute__((ext_vector_type(16))) __bf16 v16bf;
typedef __attribute__((ext_vector_type(8)))  float  v8f;

constexpr int CB = 4;      // batch
constexpr int CC = 128;    // channels
constexpr int CH = 128;    // height
constexpr int CW = 128;    // width
constexpr int MT = 32;     // columns per workgroup
constexpr int LDK = 132;   // bf16 stride of [n][k] LDS tiles (even, padded)
constexpr int LDY = 133;   // f32 stride of gate tile (odd, padded)
constexpr int NTHREADS = 256;   // 8 waves of 32
constexpr int PFD = 4;     // cache-prefetch distance (steps)

union BFrag { v16bf v; unsigned u[8]; };

// Load a 32x16 (KxN) bf16 B-fragment from an LDS tile stored as [n][k].
// ISA 7.12.2: lane l -> N = l%16 (+16*t), VGPR i -> K = (i<4 ? 2i : 8+2i) + 8*(l/16) (+32*kk).
__device__ __forceinline__ v16bf load_bfrag(const bf16_t* tile, int kk, int t,
                                            int hi, int lr) {
  BFrag f;
  const unsigned* p = (const unsigned*)tile;
  const int base = (t * 16 + lr) * LDK + kk * 32 + hi * 8;  // bf16 units, even
#pragma unroll
  for (int i = 0; i < 8; ++i) f.u[i] = p[(base >> 1) + ((i < 4) ? i : 4 + i)];
  return f.v;
}

// Build the 16x32 bf16 A-fragment (ISA 7.12.2 16-bit A layout) for K-tile kk
// directly from a global f32 row-major [out][in] weight matrix.
__device__ __forceinline__ void build_afrag(const float* __restrict__ w,
                                            int arow, int hi, BFrag* fr) {
#pragma unroll
  for (int kk = 0; kk < 4; ++kk) {
#pragma unroll
    for (int i = 0; i < 8; ++i) {
      const int kb = ((i < 4) ? 2 * i : 8 + 2 * i) + 8 * hi + kk * 32;
      union { bf16_t h[2]; unsigned u; } pk;
      pk.h[0] = (bf16_t)w[arow * CC + kb];
      pk.h[1] = (bf16_t)w[arow * CC + kb + 1];
      fr[kk].u[i] = pk.u;
    }
  }
}

// ---------------------------------------------------------------------------
// Phase A: south (fwd) and north (bwd) scans over H.  Input x/y in [B,C,H,W].
// Output written TRANSPOSED as [B,C,W,H] so phase B reads are coalesced.
// ---------------------------------------------------------------------------
__global__ __launch_bounds__(NTHREADS)
void scan_rows_kernel(const float* __restrict__ x, const float* __restrict__ y,
                      const float* __restrict__ w1,  const float* __restrict__ b1,
                      const float* __restrict__ w2,  const float* __restrict__ b2,
                      const float* __restrict__ w9,  const float* __restrict__ b9,
                      const float* __restrict__ w10, const float* __restrict__ b10,
                      float* __restrict__ hsT, float* __restrict__ hnT) {
  const int dir = blockIdx.y;                 // 0 = south, 1 = north
  const float* wA = dir ? w9  : w1;  const float* bA = dir ? b9  : b1;
  const float* wB = dir ? w10 : w2;  const float* bB = dir ? b10 : b2;
  float* outT = dir ? hnT : hsT;
  const bool fwd = (dir == 0);
  const bool relu_first = (dir == 1);

  const int bidx = blockIdx.x >> 2;
  const int w0 = (blockIdx.x & 3) * MT;

  __shared__ bf16_t sTX[MT * LDK];   // x_k tile, [n][c] bf16
  __shared__ bf16_t sTH[MT * LDK];   // h_{prev} tile, [n][c] bf16
  __shared__ float  sY [MT * LDY];   // gate tile, [n][c] f32

  const int tid = threadIdx.x;
  const int lane = tid & 31, wv = tid >> 5;
  const int hi = lane >> 4, lr = lane & 15;
  const int rowbase = wv * 16;
  const int arow = rowbase + lr;

  // Loop-invariant A-fragments (weights) in registers for the whole scan.
  BFrag fa[4], fb[4];
  build_afrag(wA, arow, hi, fa);
  build_afrag(wB, arow, hi, fb);

  // Per-lane bias registers matching the f32 C/D layout (M = rowbase+8*hi+v).
  float rBA[8], rBB[8];
#pragma unroll
  for (int v = 0; v < 8; ++v) {
    const int cc = rowbase + 8 * hi + v;
    rBA[v] = bA[cc];
    rBB[v] = bB[cc];
  }

  const int sn = tid & 31;       // staging column (coalesced on fastest dim)
  const int scrow = tid >> 5;    // staging row group

  // Software-pipeline registers: next step's x/y tile rows for this thread.
  float px[16], py[16];

  // ---- step 0: h_0 = (relu?) x slice, no matmul ----
  {
    const int k = fwd ? 0 : (CH - 1);
#pragma unroll
    for (int r = 0; r < 16; ++r) {
      const int c = scrow * 16 + r;
      float xv = x[((bidx * CC + c) * CH + k) * CW + w0 + sn];
      float h0 = relu_first ? fmaxf(xv, 0.f) : xv;
      sTH[sn * LDK + c] = (bf16_t)h0;
      outT[((bidx * CC + c) * CW + (w0 + sn)) * CH + k] = h0;
    }
    // prefetch step 1 tiles into registers (overlaps with barrier)
    const int k1 = fwd ? 1 : (CH - 2);
    const int kg1 = fwd ? (k1 - 1) : (k1 + 1);
#pragma unroll
    for (int r = 0; r < 16; ++r) {
      const int c = scrow * 16 + r;
      px[r] = x[((bidx * CC + c) * CH + k1) * CW + w0 + sn];
      py[r] = y[((bidx * CC + c) * CH + kg1) * CW + w0 + sn];
    }
    __syncthreads();
  }

  for (int s = 1; s < CH; ++s) {
    const int k = fwd ? s : (CH - 1 - s);

    // Commit pipelined registers to LDS ([n][c]).
#pragma unroll
    for (int r = 0; r < 16; ++r) {
      const int c = scrow * 16 + r;
      sTX[sn * LDK + c] = (bf16_t)px[r];
      sY [sn * LDY + c] = py[r];
    }
    __syncthreads();

    // Issue next step's global loads now; they complete during the WMMA work.
    if (s + 1 < CH) {
      const int kn = fwd ? (s + 1) : (CH - 2 - s);
      const int kgn = fwd ? (kn - 1) : (kn + 1);
#pragma unroll
      for (int r = 0; r < 16; ++r) {
        const int c = scrow * 16 + r;
        px[r] = x[((bidx * CC + c) * CH + kn) * CW + w0 + sn];
        py[r] = y[((bidx * CC + c) * CH + kgn) * CW + w0 + sn];
      }
    }
    // Cache prefetch further ahead (no counters, warms WGP$/L2).
    if (s + PFD < CH) {
      const int kp = fwd ? (s + PFD) : (CH - 1 - s - PFD);
#pragma unroll
      for (int r = 0; r < 16; ++r) {
        const int c = scrow * 16 + r;
        __builtin_prefetch(&x[((bidx * CC + c) * CH + kp) * CW + w0 + sn], 0, 1);
      }
    }

    // Pull gate into registers so only two barriers per step are needed.
    float gv[2][8];
#pragma unroll
    for (int t = 0; t < 2; ++t) {
      const int n = t * 16 + lr;
#pragma unroll
      for (int v = 0; v < 8; ++v) gv[t][v] = sY[n * LDY + rowbase + 8 * hi + v];
    }

    v8f accA[2] = {}, accB[2] = {};
#pragma unroll
    for (int kk = 0; kk < 4; ++kk) {
#pragma unroll
      for (int t = 0; t < 2; ++t) {
        v16bf bx = load_bfrag(sTX, kk, t, hi, lr);
        v16bf bh = load_bfrag(sTH, kk, t, hi, lr);
        accA[t] = __builtin_amdgcn_wmma_f32_16x16x32_bf16(
            false, fa[kk].v, false, bx, (short)0, accA[t], false, false);
        accB[t] = __builtin_amdgcn_wmma_f32_16x16x32_bf16(
            false, fb[kk].v, false, bh, (short)0, accB[t], false, false);
      }
    }
    __syncthreads();   // all B-fragment reads done before sTH overwritten

    // h = relu(wa*x + ba + (wb*h_prev + bb) * gate); write back + store.
#pragma unroll
    for (int t = 0; t < 2; ++t) {
      const int n = t * 16 + lr;
#pragma unroll
      for (int v = 0; v < 8; ++v) {
        const int cc = rowbase + 8 * hi + v;
        float h = fmaxf(accA[t][v] + rBA[v] + (accB[t][v] + rBB[v]) * gv[t][v],
                        0.f);
        sTH[n * LDK + cc] = (bf16_t)h;
        outT[((bidx * CC + cc) * CW + (w0 + n)) * CH + k] = h;
      }
    }
    // next iteration's staging barrier orders sTH writes vs. next reads
  }
}

// ---------------------------------------------------------------------------
// Phase B: 4 diagonal scans over W.  Inputs hsT/hnT and yT in [B,C,W,H]
// (coalesced step slices).  Results accumulated atomically into outT [B,C,W,H].
// ---------------------------------------------------------------------------
__global__ __launch_bounds__(NTHREADS)
void scan_cols_kernel(const float* __restrict__ hsT, const float* __restrict__ hnT,
                      const float* __restrict__ yT,
                      const float* __restrict__ w4,  const float* __restrict__ b4,
                      const float* __restrict__ w5,  const float* __restrict__ b5,
                      const float* __restrict__ w7,  const float* __restrict__ b7,
                      const float* __restrict__ w8,  const float* __restrict__ b8,
                      const float* __restrict__ w12, const float* __restrict__ b12,
                      const float* __restrict__ w13, const float* __restrict__ b13,
                      const float* __restrict__ w15, const float* __restrict__ b15,
                      const float* __restrict__ w16, const float* __restrict__ b16,
                      float* __restrict__ outT) {
  const int scan = blockIdx.y;   // 0=SE 1=SW 2=NE 3=NW
  const float* src = (scan < 2) ? hsT : hnT;
  const float* wA = (scan == 0) ? w4 : (scan == 1) ? w7 : (scan == 2) ? w12 : w15;
  const float* bA = (scan == 0) ? b4 : (scan == 1) ? b7 : (scan == 2) ? b12 : b15;
  const float* wB = (scan == 0) ? w5 : (scan == 1) ? w8 : (scan == 2) ? w13 : w16;
  const float* bB = (scan == 0) ? b5 : (scan == 1) ? b8 : (scan == 2) ? b13 : b16;
  const bool fwd = ((scan & 1) == 0);

  const int bidx = blockIdx.x >> 2;
  const int hb0 = (blockIdx.x & 3) * MT;

  __shared__ bf16_t sTX[MT * LDK];
  __shared__ bf16_t sTH[MT * LDK];
  __shared__ float  sY [MT * LDY];

  const int tid = threadIdx.x;
  const int lane = tid & 31, wv = tid >> 5;
  const int hi = lane >> 4, lr = lane & 15;
  const int rowbase = wv * 16;
  const int arow = rowbase + lr;

  BFrag fa[4], fb[4];
  build_afrag(wA, arow, hi, fa);
  build_afrag(wB, arow, hi, fb);

  float rBA[8], rBB[8];
#pragma unroll
  for (int v = 0; v < 8; ++v) {
    const int cc = rowbase + 8 * hi + v;
    rBA[v] = bA[cc];
    rBB[v] = bB[cc];
  }

  const int sn = tid & 31;
  const int scrow = tid >> 5;

  float px[16], py[16];

  // ---- step 0: h_0 = relu(src column), contributes to the sum ----
  {
    const int w = fwd ? 0 : (CW - 1);
#pragma unroll
    for (int r = 0; r < 16; ++r) {
      const int c = scrow * 16 + r;
      float v0 = src[((bidx * CC + c) * CW + w) * CH + hb0 + sn];
      float h0 = fmaxf(v0, 0.f);
      sTH[sn * LDK + c] = (bf16_t)h0;
      atomicAdd(&outT[((bidx * CC + c) * CW + w) * CH + hb0 + sn], h0);
    }
    const int w1i = fwd ? 1 : (CW - 2);
    const int wg1 = fwd ? (w1i - 1) : (w1i + 1);
#pragma unroll
    for (int r = 0; r < 16; ++r) {
      const int c = scrow * 16 + r;
      px[r] = src[((bidx * CC + c) * CW + w1i) * CH + hb0 + sn];
      py[r] = yT[((bidx * CC + c) * CW + wg1) * CH + hb0 + sn];
    }
    __syncthreads();
  }

  for (int s = 1; s < CW; ++s) {
    const int w = fwd ? s : (CW - 1 - s);

#pragma unroll
    for (int r = 0; r < 16; ++r) {
      const int c = scrow * 16 + r;
      sTX[sn * LDK + c] = (bf16_t)px[r];
      sY [sn * LDY + c] = py[r];
    }
    __syncthreads();

    if (s + 1 < CW) {
      const int wn = fwd ? (s + 1) : (CW - 2 - s);
      const int wgn = fwd ? (wn - 1) : (wn + 1);
#pragma unroll
      for (int r = 0; r < 16; ++r) {
        const int c = scrow * 16 + r;
        px[r] = src[((bidx * CC + c) * CW + wn) * CH + hb0 + sn];
        py[r] = yT[((bidx * CC + c) * CW + wgn) * CH + hb0 + sn];
      }
    }
    if (s + PFD < CW) {
      const int wp = fwd ? (s + PFD) : (CW - 1 - s - PFD);
#pragma unroll
      for (int r = 0; r < 16; ++r) {
        const int c = scrow * 16 + r;
        __builtin_prefetch(&src[((bidx * CC + c) * CW + wp) * CH + hb0 + sn],
                           0, 1);
      }
    }

    float gv[2][8];
#pragma unroll
    for (int t = 0; t < 2; ++t) {
      const int n = t * 16 + lr;
#pragma unroll
      for (int v = 0; v < 8; ++v) gv[t][v] = sY[n * LDY + rowbase + 8 * hi + v];
    }

    v8f accA[2] = {}, accB[2] = {};
#pragma unroll
    for (int kk = 0; kk < 4; ++kk) {
#pragma unroll
      for (int t = 0; t < 2; ++t) {
        v16bf bx = load_bfrag(sTX, kk, t, hi, lr);
        v16bf bh = load_bfrag(sTH, kk, t, hi, lr);
        accA[t] = __builtin_amdgcn_wmma_f32_16x16x32_bf16(
            false, fa[kk].v, false, bx, (short)0, accA[t], false, false);
        accB[t] = __builtin_amdgcn_wmma_f32_16x16x32_bf16(
            false, fb[kk].v, false, bh, (short)0, accB[t], false, false);
      }
    }
    __syncthreads();

#pragma unroll
    for (int t = 0; t < 2; ++t) {
      const int n = t * 16 + lr;
#pragma unroll
      for (int v = 0; v < 8; ++v) {
        const int cc = rowbase + 8 * hi + v;
        float h = fmaxf(accA[t][v] + rBA[v] + (accB[t][v] + rBB[v]) * gv[t][v],
                        0.f);
        sTH[n * LDK + cc] = (bf16_t)h;
        atomicAdd(&outT[((bidx * CC + cc) * CW + w) * CH + hb0 + n], h);
      }
    }
  }
}

// ---------------------------------------------------------------------------
// Helpers: zero-fill + tiled HW transpose ([R][Cc] -> [Cc][R] per plane).
// ---------------------------------------------------------------------------
__global__ void zero_kernel(float* __restrict__ p, int n) {
  int i = blockIdx.x * blockDim.x + threadIdx.x;
  if (i < n) p[i] = 0.f;
}

__global__ void transpose_hw_kernel(const float* __restrict__ in,
                                    float* __restrict__ out, int R, int Cc) {
  __shared__ float tile[32][33];
  const int plane = blockIdx.z;
  const float* ip = in + (size_t)plane * R * Cc;
  float* op = out + (size_t)plane * R * Cc;
  const int r0 = blockIdx.y * 32, c0 = blockIdx.x * 32;
  const int tx = threadIdx.x, ty = threadIdx.y;  // 32 x 8
#pragma unroll
  for (int i = 0; i < 32; i += 8)
    tile[ty + i][tx] = ip[(size_t)(r0 + ty + i) * Cc + c0 + tx];
  __syncthreads();
#pragma unroll
  for (int i = 0; i < 32; i += 8)
    op[(size_t)(c0 + ty + i) * R + r0 + tx] = tile[tx][ty + i];
}

// ---------------------------------------------------------------------------
extern "C" void kernel_launch(void* const* d_in, const int* in_sizes, int n_in,
                              void* d_out, int out_size, void* d_ws,
                              size_t ws_size, hipStream_t stream) {
  const float* x = (const float*)d_in[0];
  const float* y = (const float*)d_in[1];
  const float* w1  = (const float*)d_in[2];  const float* b1  = (const float*)d_in[3];
  const float* w2  = (const float*)d_in[4];  const float* b2  = (const float*)d_in[5];
  const float* w4  = (const float*)d_in[6];  const float* b4  = (const float*)d_in[7];
  const float* w5  = (const float*)d_in[8];  const float* b5  = (const float*)d_in[9];
  const float* w7  = (const float*)d_in[10]; const float* b7  = (const float*)d_in[11];
  const float* w8  = (const float*)d_in[12]; const float* b8  = (const float*)d_in[13];
  const float* w9  = (const float*)d_in[14]; const float* b9  = (const float*)d_in[15];
  const float* w10 = (const float*)d_in[16]; const float* b10 = (const float*)d_in[17];
  const float* w12 = (const float*)d_in[18]; const float* b12 = (const float*)d_in[19];
  const float* w13 = (const float*)d_in[20]; const float* b13 = (const float*)d_in[21];
  const float* w15 = (const float*)d_in[22]; const float* b15 = (const float*)d_in[23];
  const float* w16 = (const float*)d_in[24]; const float* b16 = (const float*)d_in[25];

  const size_t N = (size_t)CB * CC * CH * CW;  // 8,388,608 elements
  char* ws = (char*)d_ws;
  float* hsT  = (float*)(ws);
  float* hnT  = (float*)(ws + 1 * N * sizeof(float));
  float* yT   = (float*)(ws + 2 * N * sizeof(float));
  float* outT = (float*)(ws + 3 * N * sizeof(float));
  float* out = (float*)d_out;

  // outT accumulator must start at zero (harness poisons ws).
  zero_kernel<<<(int)((N + 255) / 256), 256, 0, stream>>>(outT, (int)N);
  // y -> yT ([B,C,W,H]) so phase-B gate reads are coalesced.
  transpose_hw_kernel<<<dim3(CW / 32, CH / 32, CB * CC), dim3(32, 8), 0,
                        stream>>>(y, yT, CH, CW);
  // Phase A: south + north scans in parallel.
  scan_rows_kernel<<<dim3(16, 2), NTHREADS, 0, stream>>>(
      x, y, w1, b1, w2, b2, w9, b9, w10, b10, hsT, hnT);
  // Phase B: 4 diagonal scans in parallel, accumulating into outT.
  scan_cols_kernel<<<dim3(16, 4), NTHREADS, 0, stream>>>(
      hsT, hnT, yT, w4, b4, w5, b5, w7, b7, w8, b8, w12, b12, w13, b13, w15,
      b15, w16, b16, outT);
  // outT [B,C,W,H] -> out [B,C,H,W].
  transpose_hw_kernel<<<dim3(CH / 32, CW / 32, CB * CC), dim3(32, 8), 0,
                        stream>>>(outT, out, CW, CH);
}
